// LengthRegulator_43894565765226
// MI455X (gfx1250) — compile-verified, matrix operand order checked by reading, not run on hardware
//
#include <hip/hip_runtime.h>
#include <hip/hip_bf16.h>

#define B_   32
#define H_   384
#define T_   1024
#define K_   3
#define KK_  (H_ * K_)     // 1152
#define MAXLEN 4096
#define XROW 1032          // padded time row (16B aligned row stride)
#define EPSF 1e-5f

// WMMA tiling
#define MT 128
#define NT 128
#define KC 32
#define NKSTEP (KK_ / KC)  // 36
#define LK 40              // LDS row stride in halfs (80B) -> conflict-free b128 reads

typedef __attribute__((ext_vector_type(16))) __bf16          v16bf;
typedef __attribute__((ext_vector_type(8)))  float           v8f;
typedef __attribute__((ext_vector_type(8)))  unsigned short  us8;
typedef __attribute__((ext_vector_type(16))) unsigned short  us16;

union FragU { us16 u; v16bf b; };

// ---- CDNA5 async global->LDS path (guarded; manual fallback keeps compile green) ----
#if defined(__HIP_DEVICE_COMPILE__) && __has_builtin(__builtin_amdgcn_global_load_async_to_lds_b128)
#define ASYNC_A 1
typedef int v4i_vs __attribute__((vector_size(16)));
// builtin signature (from compiler diagnostic): arg0 = generic `int __vector(4)*`,
// arg1 = LDS-side pointer; cast via void* and C-style addrspace conversion.
#define ASYNC_GSRC(p) ((v4i_vs*)(void*)(p))
#define ASYNC_LDST(p) ((__attribute__((address_space(3))) v4i_vs*)(v4i_vs*)(void*)(p))
#if __has_builtin(__builtin_amdgcn_s_wait_asynccnt)
#define WAIT_ASYNC() __builtin_amdgcn_s_wait_asynccnt(0)
#else
#define WAIT_ASYNC() asm volatile("s_wait_asynccnt 0" ::: "memory")
#endif
#else
#define ASYNC_A 0
#define WAIT_ASYNC() ((void)0)
#endif

__device__ __forceinline__ unsigned short f2bf(float f) {
    unsigned u = __float_as_uint(f);
    unsigned r = 0x7FFFu + ((u >> 16) & 1u);   // round-to-nearest-even
    return (unsigned short)((u + r) >> 16);
}

// ---- pack W [H,H,K] f32 -> Wr bf16 [o][k*384 + i] ----
__global__ void pack_w_kernel(const float* __restrict__ W, unsigned short* __restrict__ Wr) {
    int i = blockIdx.x * blockDim.x + threadIdx.x;
    if (i >= H_ * H_ * K_) return;
    int o   = i / KK_;
    int rem = i - o * KK_;
    int k   = rem / H_;
    int c   = rem - k * H_;
    Wr[i] = f2bf(W[(o * H_ + c) * K_ + k]);
}

// ---- pack x f32 [B,H,T] -> bf16 padded [B,H,XROW], valid cols [1..T] ----
__global__ void pack_x_kernel(const float* __restrict__ X, unsigned short* __restrict__ Xp) {
    int i = blockIdx.x * blockDim.x + threadIdx.x;
    if (i >= B_ * H_ * XROW) return;
    int bh = i / XROW;
    int c  = i - bh * XROW;
    float v = (c >= 1 && c <= T_) ? X[(size_t)bh * T_ + (c - 1)] : 0.f;
    Xp[i] = f2bf(v);
}

// ---- zero the 8 pad columns of hpad (col 0 and 1025..1031) ----
__global__ void zero_border_kernel(unsigned short* __restrict__ Hp) {
    int i = blockIdx.x * blockDim.x + threadIdx.x;
    if (i >= B_ * H_ * 8) return;
    int bh = i >> 3;
    int j  = i & 7;
    int c  = (j == 0) ? 0 : (T_ + j);
    Hp[(size_t)bh * XROW + c] = 0;
}

// ---- bf16 WMMA conv-as-GEMM, double-buffered LDS pipeline (1 barrier / K-step) ----
__global__ __launch_bounds__(256)
void conv_wmma_kernel(const unsigned short* __restrict__ Wr,   // [H][1152] bf16
                      const unsigned short* __restrict__ Xp,   // [B][H][XROW] bf16
                      const float* __restrict__ bias,          // [H]
                      float* __restrict__ Y)                   // [B][H][T] f32
{
    __shared__ unsigned short As[2][MT * LK];   // weights [m][k]
    __shared__ unsigned short Bs[2][NT * LK];   // activations transposed [n][k]

    const int t0  = blockIdx.x * NT;
    const int m0  = blockIdx.y * MT;
    const int b   = blockIdx.z;
    const int tid = threadIdx.x;
    const int lane = tid & 31, wid = tid >> 5;
    const int wm = wid >> 2, wn = wid & 3;   // 2x4 wave grid

    v8f acc[4][2] = {};

    const int ar  = tid >> 1;            // A-fill: row, 2 threads/row
    const int as  = (tid & 1) * 16;      //         half-row (16 halfs)
    const int bk  = tid >> 3;            // B-fill: k row 0..31
    const int bn0 = tid & 7;             //         n phase 0..7

    auto fill = [&](int buf, int kk0) {
        const int tap = kk0 / H_;        // 384 % 32 == 0 -> whole chunk in one tap
        const int i0  = kk0 - tap * H_;
        // stage A tile (128x32): 2x b128 per thread
        const unsigned short* ga = Wr + (size_t)(m0 + ar) * KK_ + kk0 + as;
        unsigned short* la = &As[buf][ar * LK + as];
#if ASYNC_A
        __builtin_amdgcn_global_load_async_to_lds_b128(ASYNC_GSRC(ga), ASYNC_LDST(la), 0, 0);
        __builtin_amdgcn_global_load_async_to_lds_b128(ASYNC_GSRC(ga), ASYNC_LDST(la), 16, 0);
#else
        {
            const us8* g = (const us8*)ga;
            us8 d0 = g[0], d1 = g[1];
            *(us8*)(la)     = d0;
            *(us8*)(la + 8) = d1;
        }
#endif
        // stage B tile transposed to [n][k] (scalar u16: 2B granularity transpose)
        const unsigned short* src = Xp + ((size_t)(b * H_ + i0 + bk)) * XROW + t0 + tap;
        #pragma unroll
        for (int jj = 0; jj < 16; ++jj) {
            int n = bn0 + jj * 8;
            Bs[buf][n * LK + bk] = src[n];
        }
        // prefetch next step's activation rows into cache
        if (kk0 + KC < KK_) {
            const int ntap = (kk0 + KC) / H_;
            const int ni0  = (kk0 + KC) - ntap * H_;
            __builtin_prefetch(Xp + ((size_t)(b * H_ + ni0 + bk)) * XROW + t0 + ntap, 0, 1);
        }
    };

    fill(0, 0);

    for (int it = 0; it < NKSTEP; ++it) {
        WAIT_ASYNC();            // my async LDS writes for buf[it&1] retired
        __syncthreads();         // everyone's writes visible; everyone done reading buf[(it+1)&1]
        if (it + 1 < NKSTEP) fill((it + 1) & 1, (it + 1) * KC);
        const int buf = it & 1;

        // fragment loads (documented 16-bit A/B lane layout: lane%16 = row,
        // lane/16 selects K-halves {0..7,16..23} vs {8..15,24..31})
        const int rr = lane & 15;
        const int kb = (lane >> 4) << 3;
        FragU afr[4], bfr[2];
        #pragma unroll
        for (int i = 0; i < 4; ++i) {
            const unsigned short* p = &As[buf][(wm * 64 + i * 16 + rr) * LK + kb];
            us8 lo = *(const us8*)p;
            us8 hi = *(const us8*)(p + 16);
            afr[i].u = __builtin_shufflevector(lo, hi, 0,1,2,3,4,5,6,7,8,9,10,11,12,13,14,15);
        }
        #pragma unroll
        for (int j = 0; j < 2; ++j) {
            const unsigned short* p = &Bs[buf][(wn * 32 + j * 16 + rr) * LK + kb];
            us8 lo = *(const us8*)p;
            us8 hi = *(const us8*)(p + 16);
            bfr[j].u = __builtin_shufflevector(lo, hi, 0,1,2,3,4,5,6,7,8,9,10,11,12,13,14,15);
        }

        #pragma unroll
        for (int i = 0; i < 4; ++i)
            #pragma unroll
            for (int j = 0; j < 2; ++j)
                acc[i][j] = __builtin_amdgcn_wmma_f32_16x16x32_bf16(
                    false, afr[i].b, false, bfr[j].b, (short)0, acc[i][j], false, false);
    }

    // epilogue: C layout — VGPR r holds M=r (lanes 0-15) / M=8+r (lanes 16-31), N=lane%16
    const int nn = lane & 15;
    const int mo = (lane >> 4) * 8;
    #pragma unroll
    for (int i = 0; i < 4; ++i) {
        const int mbase = m0 + wm * 64 + i * 16 + mo;
        #pragma unroll
        for (int j = 0; j < 2; ++j) {
            const int n = t0 + wn * 32 + j * 16 + nn;
            #pragma unroll
            for (int r = 0; r < 8; ++r) {
                const int m = mbase + r;
                Y[((size_t)(b * H_ + m)) * T_ + n] = acc[i][j][r] + bias[m];
            }
        }
    }
}

// ---- LayerNorm over H + ReLU, write bf16 into padded buffer (lane owns one t -> coalesced) ----
__global__ void ln_relu_pack_kernel(const float* __restrict__ Y,
                                    const float* __restrict__ g,
                                    const float* __restrict__ be,
                                    unsigned short* __restrict__ Hp) {
    const int t = blockIdx.x * blockDim.x + threadIdx.x;
    const int b = blockIdx.y;
    const float* y = Y + (size_t)b * H_ * T_ + t;
    float s = 0.f, s2 = 0.f;
    for (int h = 0; h < H_; ++h) { float v = y[(size_t)h * T_]; s += v; s2 += v * v; }
    const float mean = s * (1.f / H_);
    const float var  = s2 * (1.f / H_) - mean * mean;
    const float rstd = rsqrtf(var + EPSF);
    unsigned short* o = Hp + (size_t)b * H_ * XROW + 1 + t;
    for (int h = 0; h < H_; ++h) {
        float v = (y[(size_t)h * T_] - mean) * rstd * g[h] + be[h];
        o[(size_t)h * XROW] = f2bf(fmaxf(v, 0.f));
    }
}

// ---- LN + ReLU + Linear(H->1) + relu + floor(exp) -> int32 durations ----
__global__ void ln_proj_kernel(const float* __restrict__ Y,
                               const float* __restrict__ g,
                               const float* __restrict__ be,
                               const float* __restrict__ Wl,
                               const float* __restrict__ bl,
                               int* __restrict__ dur) {
    const int t = blockIdx.x * blockDim.x + threadIdx.x;
    const int b = blockIdx.y;
    const float* y = Y + (size_t)b * H_ * T_ + t;
    float s = 0.f, s2 = 0.f;
    for (int h = 0; h < H_; ++h) { float v = y[(size_t)h * T_]; s += v; s2 += v * v; }
    const float mean = s * (1.f / H_);
    const float var  = s2 * (1.f / H_) - mean * mean;
    const float rstd = rsqrtf(var + EPSF);
    float z = 0.f;
    for (int h = 0; h < H_; ++h) {
        float v = (y[(size_t)h * T_] - mean) * rstd * g[h] + be[h];
        z += fmaxf(v, 0.f) * Wl[h];
    }
    z = fmaxf(z + bl[0], 0.f);
    dur[b * T_ + t] = (int)floorf(expf(z));
}

// ---- per-batch inclusive cumsum (one lane per batch) ----
__global__ void cumsum_kernel(const int* __restrict__ dur, int* __restrict__ cum) {
    int b = threadIdx.x;
    if (b >= B_) return;
    int run = 0;
    for (int t = 0; t < T_; ++t) { run += dur[b * T_ + t]; cum[b * T_ + t] = run; }
}

// ---- binary-search gather: aligned[b,h,pos] = x[b,h,min(idx,T-1)] * valid ----
__global__ void gather_kernel(const float* __restrict__ X,
                              const int* __restrict__ cum,
                              float* __restrict__ out) {
    const int pos = blockIdx.x * blockDim.x + threadIdx.x;
    const int b   = blockIdx.y;
    const int* c  = cum + b * T_;
    const int total = c[T_ - 1];
    int lo = 0, hi = T_;
    while (lo < hi) { int mid = (lo + hi) >> 1; if (c[mid] <= pos) lo = mid + 1; else hi = mid; }
    const int idx = lo < (T_ - 1) ? lo : (T_ - 1);
    const bool valid = pos < total;
    const float* xb = X + (size_t)b * H_ * T_;
    float* ob = out + (size_t)b * H_ * MAXLEN;
    for (int h = 0; h < H_; ++h)
        ob[(size_t)h * MAXLEN + pos] = valid ? xb[(size_t)h * T_ + idx] : 0.f;
}

extern "C" void kernel_launch(void* const* d_in, const int* in_sizes, int n_in,
                              void* d_out, int out_size, void* d_ws, size_t ws_size,
                              hipStream_t stream) {
    const float* x   = (const float*)d_in[0];
    const float* W1  = (const float*)d_in[1];
    const float* b1  = (const float*)d_in[2];
    const float* g1  = (const float*)d_in[3];
    const float* be1 = (const float*)d_in[4];
    const float* W2  = (const float*)d_in[5];
    const float* b2  = (const float*)d_in[6];
    const float* g2  = (const float*)d_in[7];
    const float* be2 = (const float*)d_in[8];
    const float* Wl  = (const float*)d_in[9];
    const float* bl  = (const float*)d_in[10];
    (void)in_sizes; (void)n_in; (void)out_size; (void)ws_size;

    // workspace carve-out (256B aligned)
    char* ws = (char*)d_ws;
    size_t off = 0;
    auto carve = [&](size_t bytes) -> void* {
        void* p = ws + off;
        off = (off + bytes + 255) & ~(size_t)255;
        return p;
    };
    unsigned short* Wr1 = (unsigned short*)carve((size_t)H_ * KK_ * 2);
    unsigned short* Wr2 = (unsigned short*)carve((size_t)H_ * KK_ * 2);
    unsigned short* Xp  = (unsigned short*)carve((size_t)B_ * H_ * XROW * 2);
    unsigned short* Hp  = (unsigned short*)carve((size_t)B_ * H_ * XROW * 2);
    float*          Y   = (float*)carve((size_t)B_ * H_ * T_ * 4);
    int*            cum = (int*)carve((size_t)B_ * T_ * 4);

    int*   dur     = (int*)d_out;                     // int32 durations [B,T]
    float* aligned = (float*)d_out + (size_t)B_ * T_; // f32 aligned [B,H,MAXLEN]

    const int TB = 256;
    pack_w_kernel<<<(H_ * H_ * K_ + TB - 1) / TB, TB, 0, stream>>>(W1, Wr1);
    pack_w_kernel<<<(H_ * H_ * K_ + TB - 1) / TB, TB, 0, stream>>>(W2, Wr2);
    pack_x_kernel<<<(B_ * H_ * XROW + TB - 1) / TB, TB, 0, stream>>>(x, Xp);
    zero_border_kernel<<<(B_ * H_ * 8 + TB - 1) / TB, TB, 0, stream>>>(Hp);

    dim3 cg(T_ / NT, H_ / MT, B_);
    conv_wmma_kernel<<<cg, TB, 0, stream>>>(Wr1, Xp, b1, Y);
    ln_relu_pack_kernel<<<dim3(T_ / TB, B_), TB, 0, stream>>>(Y, g1, be1, Hp);
    conv_wmma_kernel<<<cg, TB, 0, stream>>>(Wr2, Hp, b2, Y);
    ln_proj_kernel<<<dim3(T_ / TB, B_), TB, 0, stream>>>(Y, g2, be2, Wl, bl, dur);
    cumsum_kernel<<<1, 32, 0, stream>>>(dur, cum);
    gather_kernel<<<dim3(MAXLEN / TB, B_), TB, 0, stream>>>(x, cum, aligned);
}
